// VectorQuantizer_5085241279051
// MI455X (gfx1250) — matrix operand order repaired and checked by reading, not compiled, not run on hardware
//
#include <hip/hip_runtime.h>
#include <hip/hip_bf16.h>

typedef _Float16 v16h __attribute__((ext_vector_type(16)));
typedef float    v8f  __attribute__((ext_vector_type(8)));

#define DIM        128
#define K_CODES    1024
#define N_ROWS     65536
#define CT_TILES   64      // 1024 codes / 16 cols per tile
#define WAVES      8
#define OUT_ELEMS  8388608 // 65536*128

// -------------------------------------------------------------------------
// Kernel A: build fp16 hi/lo codebook B-fragments (WMMA 32x16 layout),
// ||e||^2 per code, and zero the histogram (must re-zero every call).
// B k order used here (contiguous, same as A): lane L holds column n = L%16; elements j=0..15
// hold k = (L>=16 ? 16 : 0) + j within the 32-wide k-chunk.
// -------------------------------------------------------------------------
__global__ __launch_bounds__(256) void vq_prep(
    const float* __restrict__ emb,
    _Float16* __restrict__ Bh, _Float16* __restrict__ Bl,
    float* __restrict__ ee, unsigned* __restrict__ counts)
{
    const int ct = blockIdx.x;     // 0..63
    const int t  = threadIdx.x;    // 0..255

    if (t < 16) counts[ct * 16 + t] = 0u;

#pragma unroll
    for (int u = 0; u < 8; ++u) {
        const int eidx = t * 8 + u;          // 0..2047 = kc*512 + lane*16 + j
        const int j    = eidx & 15;
        const int lane = (eidx >> 4) & 31;
        const int kc   = eidx >> 9;
        const int n    = ct * 16 + (lane & 15);
        const int k    = kc * 32 + ((lane & 16) ? 16 : 0) + j;
        const float e  = emb[n * DIM + k];
        const _Float16 h = (_Float16)e;
        Bh[ct * 2048 + eidx] = h;
        Bl[ct * 2048 + eidx] = (_Float16)(e - (float)h);
    }

    if (t < 16) {
        const int n = ct * 16 + t;
        float s = 0.f;
        for (int k = 0; k < DIM; ++k) { const float v = emb[n * DIM + k]; s += v * v; }
        ee[n] = s;
    }
}

// -------------------------------------------------------------------------
// Kernel B: per wave = one 16-row tile. WMMA distances vs all 1024 codes,
// argmin, gather winning code rows to out, squared-error partials, histogram.
// B fragments are staged global->LDS with CDNA5 async-to-LDS loads
// (ASYNCcnt-tracked, no VGPR staging), double-buffered across column tiles.
// -------------------------------------------------------------------------
__global__ __launch_bounds__(256) void vq_main(
    const float* __restrict__ x, const float* __restrict__ emb,
    const _Float16* __restrict__ Bh, const _Float16* __restrict__ Bl,
    const float* __restrict__ ee,
    float* __restrict__ outq, unsigned* __restrict__ counts,
    float* __restrict__ partials)
{
    __shared__ __align__(32) _Float16 sB[2][2][2048]; // [buf][hi/lo][kc*512+lane*16+j]
    __shared__ float sEE[2][16];
    __shared__ float wErr[WAVES];

    const int t      = threadIdx.x;
    const int wave   = t >> 5;
    const int lane   = t & 31;
    const int nlane  = lane & 15;
    const int halfId = lane >> 4;

    const int  tile    = blockIdx.x * WAVES + wave;   // 0..4095
    const long rowBase = (long)tile * 16;

    // ---- Build A fragments (fp16 hi/lo) for this wave's 16x128 tile ----
    // A k order used here (contiguous, same as B): lane L holds row m = L%16; elements j=0..15 ->
    // k = kb+j, kb = (L>=16 ? 16 : 0).
    v16h ah[4], al[4];
    {
        const float* xr = x + (rowBase + nlane) * DIM;
        const int kb = halfId * 16;
#pragma unroll
        for (int kc = 0; kc < 4; ++kc) {
            const float* p0 = xr + kc * 32 + kb;
            const float* p1 = p0 + 8;
#pragma unroll
            for (int j = 0; j < 8; ++j) {
                const float v0 = p0[j];
                const _Float16 h0 = (_Float16)v0;
                ah[kc][j]     = h0;
                al[kc][j]     = (_Float16)(v0 - (float)h0);
                const float v1 = p1[j];
                const _Float16 h1 = (_Float16)v1;
                ah[kc][j + 8] = h1;
                al[kc][j + 8] = (_Float16)(v1 - (float)h1);
            }
        }
    }

    float minv[8];
    int   mini[8];
#pragma unroll
    for (int e = 0; e < 8; ++e) { minv[e] = 3.4e38f; mini[e] = 0; }

    // Cooperative double-buffered stage of one column-tile of B fragments
    // using CDNA5 async global->LDS copies (16B per thread per array).
    // Flat generic address of an LDS object: addr[31:0] is the LDS address
    // (ISA 10.2 aperture mapping), so truncation gives the VDST operand.
    auto loadTileAsync = [&](int ct, int buf) {
        const unsigned long long gh =
            (unsigned long long)(uintptr_t)(Bh + ct * 2048 + t * 8);
        const unsigned long long gl =
            (unsigned long long)(uintptr_t)(Bl + ct * 2048 + t * 8);
        const unsigned dh = (unsigned)(uintptr_t)(const void*)&sB[buf][0][t * 8];
        const unsigned dl = (unsigned)(uintptr_t)(const void*)&sB[buf][1][t * 8];
        asm volatile("global_load_async_to_lds_b128 %0, %1, off"
                     :: "v"(dh), "v"(gh) : "memory");
        asm volatile("global_load_async_to_lds_b128 %0, %1, off"
                     :: "v"(dl), "v"(gl) : "memory");
        if (t < 16) sEE[buf][t] = ee[ct * 16 + t];
    };

    loadTileAsync(0, 0);
    asm volatile("s_wait_asynccnt 0" ::: "memory");
    __syncthreads();

    for (int ct = 0; ct < CT_TILES; ++ct) {
        const int buf = ct & 1;
        if (ct + 1 < CT_TILES) loadTileAsync(ct + 1, buf ^ 1);

        const float eecol = sEE[buf][nlane];
        v8f acc = {};
#pragma unroll
        for (int kc = 0; kc < 4; ++kc) {
            const v16h bh = *(const v16h*)&sB[buf][0][(kc * 32 + lane) * 16];
            const v16h bl = *(const v16h*)&sB[buf][1][(kc * 32 + lane) * 16];
            acc = __builtin_amdgcn_wmma_f32_16x16x32_f16(false, ah[kc], false, bh,
                                                         (short)0, acc, false, false);
            acc = __builtin_amdgcn_wmma_f32_16x16x32_f16(false, al[kc], false, bh,
                                                         (short)0, acc, false, false);
            acc = __builtin_amdgcn_wmma_f32_16x16x32_f16(false, ah[kc], false, bl,
                                                         (short)0, acc, false, false);
        }

        const int col = ct * 16 + nlane;
#pragma unroll
        for (int e = 0; e < 8; ++e) {
            const float d = fmaf(-2.f, acc[e], eecol);   // ||e||^2 - 2 x.e
            if (d < minv[e]) { minv[e] = d; mini[e] = col; }
        }
        // Producer-side completion of the async prefetch, then block barrier
        // so the next iteration's buffer is visible to all waves.
        asm volatile("s_wait_asynccnt 0" ::: "memory");
        __syncthreads();
    }

    // Argmin reduction across the 16 lanes of each half-wave (C layout:
    // lanes 0-15 hold rows 0-7, lanes 16-31 hold rows 8-15). XOR masks <=8
    // never cross the 16-lane boundary.
#pragma unroll
    for (int m = 8; m >= 1; m >>= 1) {
#pragma unroll
        for (int e = 0; e < 8; ++e) {
            const float ov = __shfl_xor(minv[e], m, 32);
            const int   oi = __shfl_xor(mini[e], m, 32);
            if (ov < minv[e] || (ov == minv[e] && oi < mini[e])) {
                minv[e] = ov; mini[e] = oi;
            }
        }
    }

    // Gather winning code rows -> out, squared-error, histogram.
    float err = 0.f;
#pragma unroll
    for (int r = 0; r < 8; ++r) {
        const long row = rowBase + halfId * 8 + r;
        const int  idx = mini[r];
        const float4* q4 = (const float4*)(emb + (long)idx * DIM + nlane * 8);
        const float4* x4 = (const float4*)(x + row * DIM + nlane * 8);
        float4* o4 = (float4*)(outq + row * DIM + nlane * 8);
        const float4 q0 = q4[0], q1 = q4[1];
        const float4 v0 = x4[0], v1 = x4[1];
        o4[0] = q0; o4[1] = q1;
        float d;
        d = q0.x - v0.x; err = fmaf(d, d, err);
        d = q0.y - v0.y; err = fmaf(d, d, err);
        d = q0.z - v0.z; err = fmaf(d, d, err);
        d = q0.w - v0.w; err = fmaf(d, d, err);
        d = q1.x - v1.x; err = fmaf(d, d, err);
        d = q1.y - v1.y; err = fmaf(d, d, err);
        d = q1.z - v1.z; err = fmaf(d, d, err);
        d = q1.w - v1.w; err = fmaf(d, d, err);
        if (nlane == 0) atomicAdd(&counts[idx], 1u);
    }

    // Deterministic error reduction: wave shuffle -> shared -> ordered sum.
#pragma unroll
    for (int m = 16; m >= 1; m >>= 1) err += __shfl_xor(err, m, 32);
    if (lane == 0) wErr[wave] = err;
    __syncthreads();
    if (t == 0) {
        float s = 0.f;
#pragma unroll
        for (int w = 0; w < WAVES; ++w) s += wErr[w];
        partials[blockIdx.x] = s;
    }
}

// -------------------------------------------------------------------------
// Kernel C: loss + perplexity (deterministic tree reductions).
// -------------------------------------------------------------------------
__global__ __launch_bounds__(1024) void vq_final(
    const unsigned* __restrict__ counts, const float* __restrict__ partials,
    float* __restrict__ out)
{
    __shared__ float red[1024];
    const int t = threadIdx.x;

    const float p = (float)counts[t] * (1.0f / 65536.0f);
    red[t] = -p * logf(p + 1e-10f);
    __syncthreads();
    for (int s = 512; s > 0; s >>= 1) {
        if (t < s) red[t] += red[t + s];
        __syncthreads();
    }
    const float H = red[0];
    __syncthreads();

    red[t] = (t < 512) ? partials[t] : 0.f;
    __syncthreads();
    for (int s = 512; s > 0; s >>= 1) {
        if (t < s) red[t] += red[t + s];
        __syncthreads();
    }
    if (t == 0) {
        const float sq = red[0];
        // q_latent_loss + 0.25*e_latent_loss, both == mean((q-x)^2)
        out[OUT_ELEMS]     = 1.25f * sq / (float)OUT_ELEMS;
        out[OUT_ELEMS + 1] = expf(H);
    }
}

extern "C" void kernel_launch(void* const* d_in, const int* in_sizes, int n_in,
                              void* d_out, int out_size, void* d_ws, size_t ws_size,
                              hipStream_t stream)
{
    const float* x   = (const float*)d_in[0];   // [16,1024,512] f32
    const float* emb = (const float*)d_in[1];   // [1024,128]    f32
    float* out = (float*)d_out;                 // 8388608 + 2 f32

    char* ws = (char*)d_ws;
    _Float16* Bh       = (_Float16*)(ws + 0);        // 262144 B
    _Float16* Bl       = (_Float16*)(ws + 262144);   // 262144 B
    float*    ee       = (float*)   (ws + 524288);   // 4096 B
    unsigned* counts   = (unsigned*)(ws + 528384);   // 4096 B
    float*    partials = (float*)   (ws + 532480);   // 2048 B

    vq_prep <<<CT_TILES,        256,  0, stream>>>(emb, Bh, Bl, ee, counts);
    vq_main <<<N_ROWS/16/WAVES, 256,  0, stream>>>(x, emb, Bh, Bl, ee, out, counts, partials);
    vq_final<<<1,               1024, 0, stream>>>(counts, partials, out);
}
